// OscGateSSM_62302795596641
// MI455X (gfx1250) — matrix operand order repaired and checked by reading, not compiled, not run on hardware
//
#include <hip/hip_runtime.h>

#define DT_C 0.01f
#define OMEGA_MAX_C 100.0f

// ---------- types matching gfx1250 WMMA builtin signatures ----------
typedef __attribute__((ext_vector_type(8)))  __bf16 v8bf;
typedef __attribute__((ext_vector_type(16))) __bf16 v16bf;
typedef __attribute__((ext_vector_type(8)))  float  v8f;

static __device__ __forceinline__ unsigned short f2bf(float f) {
  unsigned u = __float_as_uint(f);
  unsigned r = u + 0x7FFFu + ((u >> 16) & 1u);   // round-to-nearest-even
  return (unsigned short)(r >> 16);
}

// A-matrix 16x32 bf16 fragment (ISA 7.12.2): lanes 0-15 -> M=lane; VGPR halves
// cover K = [hf*8, hf*8+7] and [16+hf*8, 16+hf*8+7].
static __device__ __forceinline__ v16bf load_a_frag(const unsigned short* __restrict__ rowptr,
                                                    int k, int hf) {
  union { v16bf v; v8bf h[2]; } u;
  u.h[0] = *(const v8bf*)(rowptr + k + hf * 8);
  u.h[1] = *(const v8bf*)(rowptr + k + 16 + hf * 8);
  return u.v;
}

// B-matrix 32x16 bf16 fragment: lane holds column N=lane&15; VGPR halves
// cover K = [hf*8, hf*8+7] and [16+hf*8, 16+hf*8+7] (same as A).
static __device__ __forceinline__ v16bf load_b_frag(const unsigned short* __restrict__ colptr,
                                                    int k, int hf) {
  union { v16bf v; v8bf h[2]; } u;
  const unsigned short* p = colptr + k + hf * 8;
  u.h[0] = *(const v8bf*)(p);
  u.h[1] = *(const v8bf*)(p + 16);
  return u.v;
}

static __device__ __forceinline__ v8f wmma_bf16(v16bf a, v16bf b, v8f c) {
  return __builtin_amdgcn_wmma_f32_16x16x32_bf16(false, a, false, b, (short)0, c, false, false);
}

// ---------- conversion / packing kernels ----------
__global__ void cvt_bf16_kernel(const float* __restrict__ src,
                                unsigned short* __restrict__ dst, int n) {
  for (int i = blockIdx.x * blockDim.x + threadIdx.x; i < n; i += gridDim.x * blockDim.x)
    dst[i] = f2bf(src[i]);
}

// pack [W_omega(64,1024); W_zeta(64,1024); W_B(128,1024)] -> wg_bf(256,1024)
__global__ void pack_wg_kernel(const float* __restrict__ Wo, const float* __restrict__ Wz,
                               const float* __restrict__ Wb, unsigned short* __restrict__ wg) {
  const int n = 256 * 1024;
  for (int i = blockIdx.x * blockDim.x + threadIdx.x; i < n; i += gridDim.x * blockDim.x) {
    int row = i >> 10, col = i & 1023;
    float v;
    if (row < 64)       v = Wo[row * 1024 + col];
    else if (row < 128) v = Wz[(row - 64) * 1024 + col];
    else                v = Wb[(row - 128) * 1024 + col];
    wg[i] = f2bf(v);
  }
}

// ---------- phase 1: gate GEMM (16384x1024 @ 1024x256) + activations ----------
// Block = 8 waves covers 32 rows x 256 cols. Wave: 2x2 of 16x16 tiles
// (2 A-frags + 2 B-frags -> 4 WMMA per K-step).
__global__ __launch_bounds__(256) void gates_kernel(
    const unsigned short* __restrict__ u_bf,   // R x 1024 bf16
    const unsigned short* __restrict__ wg_bf,  // 256 x 1024 bf16
    const float* __restrict__ b_omega, const float* __restrict__ b_zeta,
    const float* __restrict__ b_B,
    float* __restrict__ A_arr, float* __restrict__ S_arr,
    float* __restrict__ d_arr, float* __restrict__ F_arr) {
  const int rb   = blockIdx.x;          // 32-row block
  const int lane = threadIdx.x & 31;
  const int wv   = threadIdx.x >> 5;    // 8 waves, each owns 2 of 16 N-tiles
  const int hf   = lane >> 4;
  const int ln   = lane & 15;
  const int ct0  = wv * 2;

  v8f acc00 = {}, acc01 = {}, acc10 = {}, acc11 = {};
  const unsigned short* arow0 = u_bf  + (size_t)(rb * 32 + ln) * 1024;
  const unsigned short* arow1 = u_bf  + (size_t)(rb * 32 + 16 + ln) * 1024;
  const unsigned short* brow0 = wg_bf + (size_t)(ct0 * 16 + ln) * 1024;
  const unsigned short* brow1 = wg_bf + (size_t)((ct0 + 1) * 16 + ln) * 1024;
  for (int k = 0; k < 1024; k += 32) {
    v16bf a0 = load_a_frag(arow0, k, hf);
    v16bf a1 = load_a_frag(arow1, k, hf);
    v16bf b0 = load_b_frag(brow0, k, hf);
    v16bf b1 = load_b_frag(brow1, k, hf);
    acc00 = wmma_bf16(a0, b0, acc00);
    acc01 = wmma_bf16(a0, b1, acc01);
    acc10 = wmma_bf16(a1, b0, acc10);
    acc11 = wmma_bf16(a1, b1, acc11);
  }

  for (int rt = 0; rt < 2; ++rt) {
    for (int ct = 0; ct < 2; ++ct) {
      v8f acc = rt ? (ct ? acc11 : acc10) : (ct ? acc01 : acc00);
      const int col = (ct0 + ct) * 16 + ln;          // 0..255
      for (int r = 0; r < 8; ++r) {
        const int row = rb * 32 + rt * 16 + r + hf * 8;  // D layout: M = r + 8*hf
        const float g = acc[r];
        if (col < 64) {                     // omega head
          float w = g + b_omega[col];
          w = (w > 20.f) ? w : logf(1.f + __expf(w));      // softplus
          w = fminf(fmaxf(w, 1e-4f), OMEGA_MAX_C);
          float Av = w * w;
          A_arr[(size_t)row * 64 + col] = Av;
          S_arr[(size_t)row * 64 + col] = 1.f / (1.f + DT_C * DT_C * Av);
        } else if (col < 128) {             // zeta head
          const int m = col - 64;
          float zt = 1.f / (1.f + __expf(-(g + b_zeta[m])));
          d_arr[(size_t)row * 64 + m] = 1.f - zt;
        } else {                            // B head -> forcing F
          const int n = col - 128;
          F_arr[(size_t)row * 128 + n] = g + b_B[n];
        }
      }
    }
  }
}

// ---------- phase 2: sequential oscillator scan (register-resident state) ----------
__global__ __launch_bounds__(64) void scan_kernel(
    const float* __restrict__ A_arr, const float* __restrict__ S_arr,
    const float* __restrict__ d_arr, const float* __restrict__ F_arr,
    unsigned short* __restrict__ xs_bf) {
  const int b = blockIdx.x;     // batch
  const int m = threadIdx.x;    // mode 0..63, fully independent 2x2 rotation
  float z = 0.f, y = 0.f;
  size_t r = (size_t)b * 4096;
  float a  = A_arr[r * 64 + m], s = S_arr[r * 64 + m], dd = d_arr[r * 64 + m];
  float f0 = F_arr[r * 128 + m], f1 = F_arr[r * 128 + 64 + m];
  for (int t = 0; t < 4096; ++t) {
    const size_t rn = r + 1;
    float a2 = 0.f, s2 = 0.f, dd2 = 0.f, f02 = 0.f, f12 = 0.f;
    if (t + 1 < 4096) {                           // prefetch next step
      a2  = A_arr[rn * 64 + m];  s2  = S_arr[rn * 64 + m];
      dd2 = d_arr[rn * 64 + m];
      f02 = F_arr[rn * 128 + m]; f12 = F_arr[rn * 128 + 64 + m];
    }
    const float zs = s * z, ys = s * y;
    const float zn = dd * (zs - DT_C * (a * ys)) + f0;
    const float yn = dd * (DT_C * zs + ys) + f1;
    xs_bf[r * 128 + m]      = f2bf(zn);
    xs_bf[r * 128 + 64 + m] = f2bf(yn);
    z = zn; y = yn;
    a = a2; s = s2; dd = dd2; f0 = f02; f1 = f12; r = rn;
  }
}

// ---------- phase 3: y = xs @ C^T + u @ D_mat^T (fused accumulators) ----------
// Block = 8 waves covers 32 rows x 256 cols; wave: 2x2 register blocking.
__global__ __launch_bounds__(256) void out_gemm_kernel(
    const unsigned short* __restrict__ u_bf,     // R x 1024
    const unsigned short* __restrict__ xs_bf,    // R x 128
    const unsigned short* __restrict__ dmat_bf,  // 1024 x 1024 (row d, col e)
    const unsigned short* __restrict__ c_bf,     // 1024 x 128  (row d, col n)
    float* __restrict__ out) {
  const int rb   = blockIdx.x;          // 32-row block
  const int cg   = blockIdx.y;          // 4 groups of 256 output cols
  const int lane = threadIdx.x & 31;
  const int wv   = threadIdx.x >> 5;
  const int hf   = lane >> 4;
  const int ln   = lane & 15;
  const int ct0  = cg * 16 + wv * 2;    // of 64 col tiles

  v8f acc00 = {}, acc01 = {}, acc10 = {}, acc11 = {};
  const unsigned short* arow0 = u_bf    + (size_t)(rb * 32 + ln) * 1024;
  const unsigned short* arow1 = u_bf    + (size_t)(rb * 32 + 16 + ln) * 1024;
  const unsigned short* brow0 = dmat_bf + (size_t)(ct0 * 16 + ln) * 1024;
  const unsigned short* brow1 = dmat_bf + (size_t)((ct0 + 1) * 16 + ln) * 1024;
  for (int k = 0; k < 1024; k += 32) {          // u @ D_mat^T (dominant GEMM)
    v16bf a0 = load_a_frag(arow0, k, hf);
    v16bf a1 = load_a_frag(arow1, k, hf);
    v16bf b0 = load_b_frag(brow0, k, hf);
    v16bf b1 = load_b_frag(brow1, k, hf);
    acc00 = wmma_bf16(a0, b0, acc00);
    acc01 = wmma_bf16(a0, b1, acc01);
    acc10 = wmma_bf16(a1, b0, acc10);
    acc11 = wmma_bf16(a1, b1, acc11);
  }
  const unsigned short* xrow0 = xs_bf + (size_t)(rb * 32 + ln) * 128;
  const unsigned short* xrow1 = xs_bf + (size_t)(rb * 32 + 16 + ln) * 128;
  const unsigned short* cb0   = c_bf  + (size_t)(ct0 * 16 + ln) * 128;
  const unsigned short* cb1   = c_bf  + (size_t)((ct0 + 1) * 16 + ln) * 128;
  for (int k = 0; k < 128; k += 32) {           // xs @ C^T
    v16bf a0 = load_a_frag(xrow0, k, hf);
    v16bf a1 = load_a_frag(xrow1, k, hf);
    v16bf b0 = load_b_frag(cb0, k, hf);
    v16bf b1 = load_b_frag(cb1, k, hf);
    acc00 = wmma_bf16(a0, b0, acc00);
    acc01 = wmma_bf16(a0, b1, acc01);
    acc10 = wmma_bf16(a1, b0, acc10);
    acc11 = wmma_bf16(a1, b1, acc11);
  }
  for (int rt = 0; rt < 2; ++rt) {
    for (int ct = 0; ct < 2; ++ct) {
      v8f acc = rt ? (ct ? acc11 : acc10) : (ct ? acc01 : acc00);
      const int col = (ct0 + ct) * 16 + ln;
      for (int r = 0; r < 8; ++r) {
        const int row = rb * 32 + rt * 16 + r + hf * 8;
        out[(size_t)row * 1024 + col] = acc[r];
      }
    }
  }
}

// ---------- host launcher ----------
extern "C" void kernel_launch(void* const* d_in, const int* in_sizes, int n_in,
                              void* d_out, int out_size, void* d_ws, size_t ws_size,
                              hipStream_t stream) {
  (void)in_sizes; (void)n_in; (void)out_size; (void)ws_size;
  const int R = 4 * 4096;   // B*T
  const int Dd = 1024, M = 64, N = 128, G = 256;

  const float* u      = (const float*)d_in[0];
  const float* W_o    = (const float*)d_in[1];
  const float* b_o    = (const float*)d_in[2];
  const float* W_z    = (const float*)d_in[3];
  const float* b_z    = (const float*)d_in[4];
  const float* W_B    = (const float*)d_in[5];
  const float* b_B    = (const float*)d_in[6];
  const float* C_in   = (const float*)d_in[7];
  const float* D_mat  = (const float*)d_in[8];
  float* out = (float*)d_out;

  // workspace layout (256B aligned)
  char* ws = (char*)d_ws;
  size_t off = 0;
  auto take = [&](size_t bytes) { char* p = ws + off; off = (off + bytes + 255) & ~(size_t)255; return p; };
  unsigned short* u_bf    = (unsigned short*)take((size_t)R * Dd * 2);
  unsigned short* wg_bf   = (unsigned short*)take((size_t)G * Dd * 2);
  unsigned short* dmat_bf = (unsigned short*)take((size_t)Dd * Dd * 2);
  unsigned short* c_bf    = (unsigned short*)take((size_t)Dd * N * 2);
  float*          A_arr   = (float*)take((size_t)R * M * 4);
  float*          S_arr   = (float*)take((size_t)R * M * 4);
  float*          d_arr   = (float*)take((size_t)R * M * 4);
  float*          F_arr   = (float*)take((size_t)R * N * 4);
  unsigned short* xs_bf   = (unsigned short*)take((size_t)R * N * 2);

  // converts
  cvt_bf16_kernel<<<4096, 256, 0, stream>>>(u, u_bf, R * Dd);
  pack_wg_kernel<<<512, 256, 0, stream>>>(W_o, W_z, W_B, wg_bf);
  cvt_bf16_kernel<<<2048, 256, 0, stream>>>(D_mat, dmat_bf, Dd * Dd);
  cvt_bf16_kernel<<<256, 256, 0, stream>>>(C_in, c_bf, Dd * N);

  // phase 1: gates (32 rows per block)
  gates_kernel<<<R / 32, 256, 0, stream>>>(u_bf, wg_bf, b_o, b_z, b_B,
                                           A_arr, S_arr, d_arr, F_arr);
  // phase 2: scan
  scan_kernel<<<4, 64, 0, stream>>>(A_arr, S_arr, d_arr, F_arr, xs_bf);

  // phase 3: output projection (32 rows x 256 cols per block)
  out_gemm_kernel<<<dim3(R / 32, 4), 256, 0, stream>>>(u_bf, xs_bf, dmat_bf, c_bf, out);
}
